// LightProtoSSM_85847806313231
// MI455X (gfx1250) — matrix operand order repaired and checked by reading, not compiled, not run on hardware
//
#include <hip/hip_runtime.h>
#include <math.h>

// ---------------------------------------------------------------------------
// MI455X / gfx1250 implementation of LightProtoSSM forward.
// All GEMMs run on the matrix cores via v_wmma_f32_16x16x32_bf16 (bf16 in,
// f32 accumulate) with double-buffered LDS tiles and vectorized f32->bf16
// staging; elementwise / scan / softmax stages are fp32 VALU.
// ---------------------------------------------------------------------------

typedef __bf16 v16bf  __attribute__((ext_vector_type(16)));
typedef __bf16 bf16x4 __attribute__((ext_vector_type(4)));
typedef float  v8f    __attribute__((ext_vector_type(8)));

constexpr int EP_BIAS = 0, EP_SOFTPLUS = 1, EP_HEAD = 2;
#define KPAD 40   // bf16 elements per LDS tile row (32 data + 8 pad)

__device__ __forceinline__ float softplus_f(float x) {
  return x > 20.f ? x : log1pf(__expf(x));
}
__device__ __forceinline__ float sigmoid_f(float x) {
  return 1.f / (1.f + __expf(-x));
}
__device__ __forceinline__ bf16x4 cvt4(float4 v) {
  bf16x4 r = { (__bf16)v.x, (__bf16)v.y, (__bf16)v.z, (__bf16)v.w };
  return r;
}

// A fragment: 16x32 bf16, ISA layout: lane L row M=L%16; lanes<16 hold
// K={0..7,16..23}, lanes>=16 hold K={8..15,24..31} (2 halfs per VGPR).
__device__ __forceinline__ v16bf load_a_frag(const __bf16* sA, int lane) {
  const int m  = lane & 15;
  const int kb = (lane >> 4) << 3;          // 0 or 8
  const __bf16* row = sA + m * KPAD;
  v16bf a;
#pragma unroll
  for (int e = 0; e < 8; ++e) a[e]     = row[kb + e];
#pragma unroll
  for (int e = 0; e < 8; ++e) a[8 + e] = row[16 + kb + e];
  return a;
}

// B fragment: 32x16 bf16, staged transposed in LDS as [n][k]. Lane L holds
// column n=L%16; lanes<16 K={0..7,16..23}, lanes>=16 K={8..15,24..31}.
__device__ __forceinline__ v16bf load_b_frag(const __bf16* sB, int lane) {
  const int n  = lane & 15;
  const int kb = (lane >> 4) << 3;          // 0 or 8
  const __bf16* row = sB + n * KPAD;
  v16bf b;
#pragma unroll
  for (int e = 0; e < 8; ++e) b[e]     = row[kb + e];
#pragma unroll
  for (int e = 0; e < 8; ++e) b[8 + e] = row[16 + kb + e];
  return b;
}

// ---------------------------------------------------------------------------
// Generic WMMA GEMM:  C[m,n] = sum_k A[m,k] * W[n,k]  (+ bias / epilogue)
// 256 threads = 8 waves arranged as WM x WN 16x16 tiles.
// Software-pipelined: tile k+1 is fetched (global_load_b128) while tile k
// feeds the WMMA from the alternate LDS buffer.
// ---------------------------------------------------------------------------
template<int WM, int WN, int EP>
__global__ void __launch_bounds__(256)
k_gemm(int M, int N, int K,
       const float* __restrict__ A, int lda,
       const float* __restrict__ W, int ldw,
       const float* __restrict__ bias,
       float* __restrict__ C, int ldc,
       const float* __restrict__ perch,
       const float* __restrict__ cbias,
       const float* __restrict__ alpha_raw,
       const float* __restrict__ ptemp)
{
  constexpr int NA4 = 16 * WM * 8;           // float4 chunks per A tile
  constexpr int NB4 = 16 * WN * 8;           // float4 chunks per B tile
  constexpr int RA  = (NA4 + 255) / 256;
  constexpr int RB  = (NB4 + 255) / 256;

  __shared__ __bf16 sA[2][16 * WM * KPAD];
  __shared__ __bf16 sB[2][16 * WN * KPAD];
  const int tid  = threadIdx.x;
  const int wave = tid >> 5, lane = tid & 31;
  const int wm = wave / WN, wn = wave % WN;
  const int rowBase = blockIdx.x * (16 * WM);
  const int colBase = blockIdx.y * (16 * WN);

  float4 ra[RA], rb[RB];

  auto fetch = [&](int k0) {
#pragma unroll
    for (int i = 0; i < RA; ++i) {
      int e = tid + i * 256;
      ra[i] = make_float4(0.f, 0.f, 0.f, 0.f);
      if (e < NA4) {
        int r = e >> 3, k4 = (e & 7) << 2;
        int m = rowBase + r;
        if (m < M) ra[i] = *(const float4*)(A + (size_t)m * lda + k0 + k4);
      }
    }
#pragma unroll
    for (int i = 0; i < RB; ++i) {
      int e = tid + i * 256;
      rb[i] = make_float4(0.f, 0.f, 0.f, 0.f);
      if (e < NB4) {
        int r = e >> 3, k4 = (e & 7) << 2;
        int n = colBase + r;
        if (n < N) rb[i] = *(const float4*)(W + (size_t)n * ldw + k0 + k4);
      }
    }
  };
  auto stage = [&](int buf) {
#pragma unroll
    for (int i = 0; i < RA; ++i) {
      int e = tid + i * 256;
      if (e < NA4) {
        int r = e >> 3, k4 = (e & 7) << 2;
        *(bf16x4*)(&sA[buf][r * KPAD + k4]) = cvt4(ra[i]);
      }
    }
#pragma unroll
    for (int i = 0; i < RB; ++i) {
      int e = tid + i * 256;
      if (e < NB4) {
        int r = e >> 3, k4 = (e & 7) << 2;
        *(bf16x4*)(&sB[buf][r * KPAD + k4]) = cvt4(rb[i]);
      }
    }
  };

  v8f acc = {0.f, 0.f, 0.f, 0.f, 0.f, 0.f, 0.f, 0.f};
  const int nt = K >> 5;

  fetch(0);
  stage(0);
  __syncthreads();
  for (int kt = 0; kt < nt; ++kt) {
    const int cur = kt & 1;
    if (kt + 1 < nt) fetch((kt + 1) << 5);         // loads in flight
    v16bf af = load_a_frag(&sA[cur][wm * 16 * KPAD], lane);
    v16bf bf = load_b_frag(&sB[cur][wn * 16 * KPAD], lane);
    acc = __builtin_amdgcn_wmma_f32_16x16x32_bf16(
        false, af, false, bf, (short)0, acc, false, false);
    if (kt + 1 < nt) stage(cur ^ 1);               // convert + ds_store_b64
    __syncthreads();
  }

  // Epilogue.  C/D layout: VGPR r -> row (r + 8*(lane>=16)), col lane%16.
  const int n     = colBase + wn * 16 + (lane & 15);
  const int mrow0 = rowBase + wm * 16 + ((lane >> 4) << 3);
  if (n < N) {
    const float bval = bias ? bias[n] : 0.f;
    float ts = 0.f, al = 0.f, cb = 0.f;
    if (EP == EP_HEAD) {
      ts = softplus_f(ptemp[0]);
      al = sigmoid_f(alpha_raw[n]);
      cb = cbias[n];
    }
#pragma unroll
    for (int r = 0; r < 8; ++r) {
      int m = mrow0 + r;
      if (m < M) {
        float v = acc[r] + bval;
        if (EP == EP_SOFTPLUS) v = softplus_f(v);
        if (EP == EP_HEAD) {
          float sim = v * ts + cb;
          v = al * sim + (1.f - al) * perch[(size_t)m * ldc + n];
        }
        C[(size_t)m * ldc + n] = v;
      }
    }
  }
}

// ---------------------------------------------------------------------------
// WMMA GEMM (N fixed = 128) fused with LayerNorm epilogue.
// PRE_RES: add residual before LN.  POST_GP: GELU then + pos[t] + meta[b].
// Block = 16 rows x 128 cols (8 waves along N), so full rows live in LDS.
// Same double-buffered pipeline as k_gemm.
// ---------------------------------------------------------------------------
template<bool PRE_RES, bool POST_GP>
__global__ void __launch_bounds__(256)
k_gemm_ln(int M, int K,
          const float* __restrict__ A, int lda,
          const float* __restrict__ W,      // [128, K]
          const float* __restrict__ bias,   // [128]
          const float* __restrict__ res,    // [M,128] (PRE_RES)
          const float* __restrict__ gamma,
          const float* __restrict__ beta,
          const float* __restrict__ pos,    // [T,128] (POST_GP)
          const float* __restrict__ meta,   // [B,128] (POST_GP)
          float* __restrict__ out, int T)
{
  constexpr int NA4 = 16 * 8;                // 128 float4 chunks (16 rows)
  constexpr int NB4 = 128 * 8;               // 1024 float4 chunks (128 rows)
  constexpr int RB  = NB4 / 256;             // 4

  __shared__ __bf16 sA[2][16 * KPAD];
  __shared__ __bf16 sB[2][128 * KPAD];
  __shared__ float  sC[16 * 128];
  __shared__ float  redS[256];
  __shared__ float  redQ[256];
  const int tid  = threadIdx.x;
  const int wave = tid >> 5, lane = tid & 31;
  const int wn = wave;
  const int rowBase = blockIdx.x * 16;

  float4 ra;            // one chunk for threads < NA4
  float4 rb[RB];

  auto fetch = [&](int k0) {
    if (tid < NA4) {
      int r = tid >> 3, k4 = (tid & 7) << 2;
      ra = *(const float4*)(A + (size_t)(rowBase + r) * lda + k0 + k4);
    }
#pragma unroll
    for (int i = 0; i < RB; ++i) {
      int e = tid + i * 256;
      int r = e >> 3, k4 = (e & 7) << 2;
      rb[i] = *(const float4*)(W + (size_t)r * K + k0 + k4);
    }
  };
  auto stage = [&](int buf) {
    if (tid < NA4) {
      int r = tid >> 3, k4 = (tid & 7) << 2;
      *(bf16x4*)(&sA[buf][r * KPAD + k4]) = cvt4(ra);
    }
#pragma unroll
    for (int i = 0; i < RB; ++i) {
      int e = tid + i * 256;
      int r = e >> 3, k4 = (e & 7) << 2;
      *(bf16x4*)(&sB[buf][r * KPAD + k4]) = cvt4(rb[i]);
    }
  };

  v8f acc = {0.f, 0.f, 0.f, 0.f, 0.f, 0.f, 0.f, 0.f};
  const int nt = K >> 5;

  fetch(0);
  stage(0);
  __syncthreads();
  for (int kt = 0; kt < nt; ++kt) {
    const int cur = kt & 1;
    if (kt + 1 < nt) fetch((kt + 1) << 5);
    v16bf af = load_a_frag(&sA[cur][0], lane);
    v16bf bf = load_b_frag(&sB[cur][wn * 16 * KPAD], lane);
    acc = __builtin_amdgcn_wmma_f32_16x16x32_bf16(
        false, af, false, bf, (short)0, acc, false, false);
    if (kt + 1 < nt) stage(cur ^ 1);
    __syncthreads();
  }

  { // scatter accumulator tile (+bias, +residual) into LDS rows
    const int f   = wn * 16 + (lane & 15);
    const int rhi = (lane >> 4) << 3;
#pragma unroll
    for (int r = 0; r < 8; ++r) {
      int rl = r + rhi;
      float v = acc[r] + bias[f];
      if (PRE_RES) v += res[(size_t)(rowBase + rl) * 128 + f];
      sC[rl * 128 + f] = v;
    }
  }
  __syncthreads();

  // LayerNorm: 16 threads per row, 8 features each.
  const int row = tid >> 4, seg = tid & 15;
  const int base = row * 128 + seg * 8;
  float s = 0.f, q = 0.f;
#pragma unroll
  for (int j = 0; j < 8; ++j) { float v = sC[base + j]; s += v; q += v * v; }
  redS[row * 16 + seg] = s;
  redQ[row * 16 + seg] = q;
  __syncthreads();
  if (seg == 0) {
    float ss = 0.f, qq = 0.f;
    for (int j = 0; j < 16; ++j) { ss += redS[row * 16 + j]; qq += redQ[row * 16 + j]; }
    float mean = ss * (1.f / 128.f);
    float var  = qq * (1.f / 128.f) - mean * mean;
    redS[row * 16] = mean;
    redQ[row * 16] = rsqrtf(var + 1e-5f);
  }
  __syncthreads();
  const float mean = redS[row * 16], inv = redQ[row * 16];
  const int m = rowBase + row;
  const int b = m / T, t = m - b * T;
#pragma unroll
  for (int j = 0; j < 8; ++j) {
    int f = seg * 8 + j;
    float v = (sC[base + j] - mean) * inv * gamma[f] + beta[f];
    if (POST_GP) {
      v = 0.5f * v * (1.f + erff(v * 0.70710678f));   // exact GELU
      v += pos[t * 128 + f] + meta[(size_t)b * 128 + f];
    }
    out[(size_t)m * 128 + f] = v;
  }
}

// ---------------------------------------------------------------------------
// meta[b,f] = meta_b[f] + site_emb[site[b]]·meta_w[f,0:16] + hour_emb[hour[b]]·meta_w[f,16:32]
// ---------------------------------------------------------------------------
__global__ void k_meta(const int* __restrict__ site_ids,
                       const int* __restrict__ hours,
                       const float* __restrict__ site_emb,
                       const float* __restrict__ hour_emb,
                       const float* __restrict__ meta_w,
                       const float* __restrict__ meta_b,
                       float* __restrict__ meta, int Bn)
{
  int idx = blockIdx.x * 256 + threadIdx.x;
  if (idx >= Bn * 128) return;
  int b = idx >> 7, f = idx & 127;
  const float* se = site_emb + site_ids[b] * 16;
  const float* he = hour_emb + hours[b] * 16;
  const float* wr = meta_w + f * 32;
  float acc = meta_b[f];
#pragma unroll
  for (int j = 0; j < 16; ++j) acc += se[j] * wr[j];
#pragma unroll
  for (int j = 0; j < 16; ++j) acc += he[j] * wr[16 + j];
  meta[idx] = acc;
}

// ---------------------------------------------------------------------------
// Depthwise causal conv (d_conv=4) + SiLU, direction-aware so the backward
// SSM never needs a materialized time-flip.
// ---------------------------------------------------------------------------
__global__ void k_conv_silu(const float* __restrict__ xin,
                            const float* __restrict__ cw,   // [128,4]
                            const float* __restrict__ cb,   // [128]
                            float* __restrict__ out,
                            int dir, int T, int total)
{
  int idx = blockIdx.x * 256 + threadIdx.x;
  if (idx >= total) return;
  int m = idx >> 7, d = idx & 127;
  int b = m / T, t = m - b * T;
  const float* w = cw + d * 4;
  float acc = cb[d];
  if (dir > 0) {
#pragma unroll
    for (int j = 0; j < 4; ++j) {
      int tt = t - 3 + j;
      if (tt >= 0) acc += xin[(size_t)(b * T + tt) * 128 + d] * w[j];
    }
  } else {
#pragma unroll
    for (int j = 0; j < 4; ++j) {
      int tt = t + 3 - j;
      if (tt < T) acc += xin[(size_t)(b * T + tt) * 128 + d] * w[j];
    }
  }
  out[idx] = acc * sigmoid_f(acc);
}

// ---------------------------------------------------------------------------
// Selective-scan: one block per batch element, one lane per channel d.
// 16-entry state in registers; B/C vectors broadcast through LDS.
// dir=+1 scans t=0..T-1 ; dir=-1 scans t=T-1..0 (outputs at native t).
// ---------------------------------------------------------------------------
__global__ void k_scan(const float* __restrict__ x,
                       const float* __restrict__ dt,
                       const float* __restrict__ Bm,
                       const float* __restrict__ Cm,
                       const float* __restrict__ A_log,  // [128,16]
                       const float* __restrict__ Dp,     // [128]
                       float* __restrict__ out, int ldo, int ocol,
                       int dir, int T)
{
  const int b = blockIdx.x, d = threadIdx.x;   // blockDim.x == 128
  float An[16];
#pragma unroll
  for (int n = 0; n < 16; ++n) An[n] = -__expf(A_log[d * 16 + n]);
  float hs[16];
#pragma unroll
  for (int n = 0; n < 16; ++n) hs[n] = 0.f;
  const float Dd = Dp[d];
  __shared__ float sB[16], sC[16];

  for (int step = 0; step < T; ++step) {
    int t = (dir > 0) ? step : (T - 1 - step);
    size_t m = (size_t)b * T + t;
    __syncthreads();
    if (d < 16)       sB[d]      = Bm[m * 16 + d];
    else if (d < 32)  sC[d - 16] = Cm[m * 16 + (d - 16)];
    __syncthreads();
    float dtv = dt[m * 128 + d];
    float xv  = x[m * 128 + d];
    float xd  = xv * dtv;
    float y = 0.f;
#pragma unroll
    for (int n = 0; n < 16; ++n) {
      float dA = __expf(An[n] * dtv);
      hs[n] = hs[n] * dA + xd * sB[n];
      y += hs[n] * sC[n];
    }
    out[m * ldo + ocol + d] = y + xv * Dd;
  }
}

// ---------------------------------------------------------------------------
// Tiny attention (T=12, 2 heads of 64): one block per batch element.
// ---------------------------------------------------------------------------
__global__ void __launch_bounds__(256)
k_attn(const float* __restrict__ qkv, float* __restrict__ o)
{
  const int b = blockIdx.x;
  __shared__ float s[2][12][12];
  const size_t base = (size_t)b * 12 * 384;

  for (int e = threadIdx.x; e < 288; e += 256) {
    int hh = e / 144, rem = e - hh * 144;
    int tq = rem / 12, tk = rem - tq * 12;
    const float* qp = qkv + base + (size_t)tq * 384 + hh * 64;
    const float* kp = qkv + base + (size_t)tk * 384 + 128 + hh * 64;
    float a = 0.f;
    for (int j = 0; j < 64; ++j) a += qp[j] * kp[j];
    s[hh][tq][tk] = a * 0.125f;                 // 1/sqrt(64)
  }
  __syncthreads();
  if (threadIdx.x < 24) {
    int hh = threadIdx.x / 12, tq = threadIdx.x % 12;
    float mx = -1e30f;
    for (int k = 0; k < 12; ++k) mx = fmaxf(mx, s[hh][tq][k]);
    float sm = 0.f;
    for (int k = 0; k < 12; ++k) { float e = __expf(s[hh][tq][k] - mx); s[hh][tq][k] = e; sm += e; }
    float inv = 1.f / sm;
    for (int k = 0; k < 12; ++k) s[hh][tq][k] *= inv;
  }
  __syncthreads();
  for (int e = threadIdx.x; e < 1536; e += 256) {
    int hh = e / 768, rem = e - hh * 768;
    int tq = rem >> 6, dd = rem & 63;
    float acc = 0.f;
    for (int tk = 0; tk < 12; ++tk)
      acc += s[hh][tq][tk] * qkv[base + (size_t)tk * 384 + 256 + hh * 64 + dd];
    o[(size_t)(b * 12 + tq) * 128 + hh * 64 + dd] = acc;
  }
}

// ---------------------------------------------------------------------------
// Row L2-normalize (128 features, clip 1e-12).  Also used for prototypes.
// ---------------------------------------------------------------------------
__global__ void k_rownorm(const float* __restrict__ in, float* __restrict__ out)
{
  __shared__ float red[128];
  const size_t m = blockIdx.x;
  float v = in[m * 128 + threadIdx.x];
  red[threadIdx.x] = v * v;
  __syncthreads();
  for (int s = 64; s > 0; s >>= 1) {
    if ((int)threadIdx.x < s) red[threadIdx.x] += red[threadIdx.x + s];
    __syncthreads();
  }
  float nrm = fmaxf(sqrtf(red[0]), 1e-12f);
  out[m * 128 + threadIdx.x] = v / nrm;
}

// ---------------------------------------------------------------------------
// Host-side orchestration.
// d_in order: emb, perch_logits, site_ids, hours, then params flattened in
// jax tree order (sorted keys; nested 'ssm' dict flattened last, sorted).
// ---------------------------------------------------------------------------
extern "C" void kernel_launch(void* const* d_in, const int* in_sizes, int n_in,
                              void* d_out, int out_size, void* d_ws, size_t ws_size,
                              hipStream_t stream) {
  (void)n_in; (void)out_size; (void)ws_size;
  const float* emb   = (const float*)d_in[0];
  const float* perch = (const float*)d_in[1];
  const int*   site  = (const int*)d_in[2];
  const int*   hrs   = (const int*)d_in[3];
  int p = 4;
  const float* attn_out_b   = (const float*)d_in[p + 0];   // [2,128]
  const float* attn_out_w   = (const float*)d_in[p + 1];   // [2,128,128]
  const float* attn_qkv_b   = (const float*)d_in[p + 2];   // [2,384]
  const float* attn_qkv_w   = (const float*)d_in[p + 3];   // [2,384,128]
  const float* class_bias   = (const float*)d_in[p + 4];   // [234]
  const float* cn_b         = (const float*)d_in[p + 5];   // [2,128]
  const float* cn_g         = (const float*)d_in[p + 6];   // [2,128]
  const float* fusion_alpha = (const float*)d_in[p + 7];   // [234]
  const float* hour_emb     = (const float*)d_in[p + 8];   // [24,16]
  const float* in_b         = (const float*)d_in[p + 9];   // [128]
  const float* in_ln_b      = (const float*)d_in[p + 10];  // [128]
  const float* in_ln_g      = (const float*)d_in[p + 11];  // [128]
  const float* in_w         = (const float*)d_in[p + 12];  // [128,1536]
  const float* merge_b      = (const float*)d_in[p + 13];  // [2,128]
  const float* merge_w      = (const float*)d_in[p + 14];  // [2,128,256]
  const float* meta_b       = (const float*)d_in[p + 15];  // [128]
  const float* meta_w       = (const float*)d_in[p + 16];  // [128,32]
  const float* norm_b       = (const float*)d_in[p + 17];  // [2,128]
  const float* norm_g       = (const float*)d_in[p + 18];  // [2,128]
  const float* pos          = (const float*)d_in[p + 19];  // [1,12,128]
  const float* proto_temp   = (const float*)d_in[p + 20];  // scalar
  const float* prototypes   = (const float*)d_in[p + 21];  // [234,128]
  const float* site_emb     = (const float*)d_in[p + 22];  // [20,16]
  const float* A_log        = (const float*)d_in[p + 23];  // [4,128,16]
  const float* B_w          = (const float*)d_in[p + 24];  // [4,16,128]
  const float* C_w          = (const float*)d_in[p + 25];  // [4,16,128]
  const float* Dpar         = (const float*)d_in[p + 26];  // [4,128]
  const float* conv_b       = (const float*)d_in[p + 27];  // [4,128]
  const float* conv_w       = (const float*)d_in[p + 28];  // [4,128,1,4]
  const float* dt_b         = (const float*)d_in[p + 29];  // [4,128]
  const float* dt_w         = (const float*)d_in[p + 30];  // [4,128,128]
  const float* in_proj      = (const float*)d_in[p + 31];  // [4,256,128]

  const int Bn = in_sizes[2];        // 4096
  const int T = 12, NC = 234;
  const int M = Bn * T;              // 49152 rows

  // workspace layout (floats)
  float* ws = (float*)d_ws;
  size_t off = 0;
  float* h      = ws + off; off += (size_t)M * 128;
  float* h2     = ws + off; off += (size_t)M * 128;
  float* tmp1   = ws + off; off += (size_t)M * 128;   // x_ssm
  float* tmp2   = ws + off; off += (size_t)M * 128;   // xc
  float* dtb    = ws + off; off += (size_t)M * 128;   // dt
  float* concat = ws + off; off += (size_t)M * 256;   // [h_f | h_b]
  float* bm     = ws + off; off += (size_t)M * 16;
  float* cm     = ws + off; off += (size_t)M * 16;
  float* metab  = ws + off; off += (size_t)Bn * 128;
  float* pn     = ws + off; off += (size_t)256 * 128;
  float* qkvb   = tmp1;     // qkv [M,384] aliases tmp1..dtb (dead then)
  float* attnb  = concat;   // attention output aliases concat (dead then)
  float* hn     = h2;       // normalized h aliases h2 at the end

  const dim3 blk(256);

  k_meta<<<dim3((Bn * 128 + 255) / 256), blk, 0, stream>>>(
      site, hrs, site_emb, hour_emb, meta_w, meta_b, metab, Bn);

  // h = gelu(LN(emb @ in_w.T + in_b)) + pos + meta
  k_gemm_ln<false, true><<<dim3(M / 16), blk, 0, stream>>>(
      M, 1536, emb, 1536, in_w, in_b, nullptr, in_ln_g, in_ln_b,
      pos, metab, h, T);

  for (int i = 0; i < 2; ++i) {
    for (int d = 0; d < 2; ++d) {            // d=0 fwd, d=1 bwd
      const int bi  = 2 * i + d;
      const int dir = (d == 0) ? 1 : -1;
      // x_ssm = h @ in_proj[bi][:128].T
      k_gemm<1, 8, EP_BIAS><<<dim3(M / 16, 1), blk, 0, stream>>>(
          M, 128, 128, h, 128, in_proj + (size_t)bi * 256 * 128, 128,
          nullptr, tmp1, 128, nullptr, nullptr, nullptr, nullptr);
      // xc = silu(depthwise causal conv)
      k_conv_silu<<<dim3((M * 128 + 255) / 256), blk, 0, stream>>>(
          tmp1, conv_w + bi * 512, conv_b + bi * 128, tmp2, dir, T, M * 128);
      // dt = softplus(xc @ dt_w.T + dt_b)
      k_gemm<1, 8, EP_SOFTPLUS><<<dim3(M / 16, 1), blk, 0, stream>>>(
          M, 128, 128, tmp2, 128, dt_w + bi * 16384, 128,
          dt_b + bi * 128, dtb, 128, nullptr, nullptr, nullptr, nullptr);
      // Bm, Cm = xc @ {B_w,C_w}.T  (N=16: waves tiled along M)
      k_gemm<8, 1, EP_BIAS><<<dim3(M / 128, 1), blk, 0, stream>>>(
          M, 16, 128, tmp2, 128, B_w + bi * 2048, 128,
          nullptr, bm, 16, nullptr, nullptr, nullptr, nullptr);
      k_gemm<8, 1, EP_BIAS><<<dim3(M / 128, 1), blk, 0, stream>>>(
          M, 16, 128, tmp2, 128, C_w + bi * 2048, 128,
          nullptr, cm, 16, nullptr, nullptr, nullptr, nullptr);
      // selective scan -> concat[:, d*128 : d*128+128]
      k_scan<<<dim3(Bn), dim3(128), 0, stream>>>(
          h, dtb, bm, cm, A_log + bi * 2048, Dpar + bi * 128,
          concat, 256, d * 128, dir, T);
    }
    // h2 = LN(concat @ merge_w.T + merge_b + h)
    k_gemm_ln<true, false><<<dim3(M / 16), blk, 0, stream>>>(
        M, 256, concat, 256, merge_w + (size_t)i * 32768, merge_b + i * 128,
        h, norm_g + i * 128, norm_b + i * 128, nullptr, nullptr, h2, T);
    // qkv = h2 @ qkv_w.T + qkv_b
    k_gemm<1, 8, EP_BIAS><<<dim3(M / 16, 3), blk, 0, stream>>>(
        M, 384, 128, h2, 128, attn_qkv_w + (size_t)i * 49152, 128,
        attn_qkv_b + i * 384, qkvb, 384, nullptr, nullptr, nullptr, nullptr);
    k_attn<<<dim3(Bn), blk, 0, stream>>>(qkvb, attnb);
    // h = LN(attnb @ out_w.T + out_b + h2)
    k_gemm_ln<true, false><<<dim3(M / 16), blk, 0, stream>>>(
        M, 128, attnb, 128, attn_out_w + (size_t)i * 16384, attn_out_b + i * 128,
        h2, cn_g + i * 128, cn_b + i * 128, nullptr, nullptr, h, T);
  }

  // prototype head: out = alpha*(hn@pn.T * softplus(temp) + class_bias) + (1-alpha)*perch
  k_rownorm<<<dim3(M), dim3(128), 0, stream>>>(h, hn);
  k_rownorm<<<dim3(NC), dim3(128), 0, stream>>>(prototypes, pn);
  k_gemm<1, 8, EP_HEAD><<<dim3(M / 16, 2), blk, 0, stream>>>(
      M, NC, 128, hn, 128, pn, 128, nullptr, (float*)d_out, NC,
      perch, class_bias, fusion_alpha, proto_temp);
}
